// Head_6184752906264
// MI455X (gfx1250) — compile-verified, matrix operand order checked by reading, not run on hardware
//
#include <hip/hip_runtime.h>
#include <hip/hip_bf16.h>

// ---------------------------------------------------------------------------
// Causal single-head attention (B=8, T=4096, C=768, d=64), bf16 WMMA path.
//   Kernel 0: prepack W{k,q,v} into bf16 WMMA B-layout fragments (one-shot).
//   Kernel 1: fused Q/K/V projection — each wave loads one x A-fragment and
//             feeds 3 WMMAs (K, Q, V). Q gets 1/sqrt(d) folded in. V stored
//             TRANSPOSED so attention P·V fragments are contiguous loads.
//   Kernel 2: FlashAttention forward, 32 keys/step: 4 WMMA (S=QK^T) +
//             4 WMMA (P·V), f32 online softmax.
// ---------------------------------------------------------------------------

typedef __attribute__((ext_vector_type(16))) __bf16 v16bf;
typedef __attribute__((ext_vector_type(8)))  __bf16 v8bf;
typedef __attribute__((ext_vector_type(8)))  float  v8f;

#define B_  8
#define T_  4096
#define C_  768
#define D_  64
#define KK_ (C_ / 32)          // 24 K-steps in the projection

static __device__ __forceinline__ v8f wmma_bf16(v16bf a, v16bf b, v8f c) {
    return __builtin_amdgcn_wmma_f32_16x16x32_bf16(
        false, a, false, b, (short)0, c, false, false);
}

// ---------------------------------------------------------------------------
// Kernel 0: prepack W into B-layout fragments.
//   wpk[(((mat*KK_ + kk)*4 + ntile)*32 + lane)*16 + i]
// Lane l<16 holds column n = ntile*16+l, c-chunks {0..7,16..23} of the K=32
// slice; lane l>=16 holds c-chunks {8..15,24..31}.  288 KB total.
// ---------------------------------------------------------------------------
__global__ __launch_bounds__(128)
void prep_w_kernel(const float* __restrict__ Wk,
                   const float* __restrict__ Wq,
                   const float* __restrict__ Wv,
                   __bf16* __restrict__ wpk)
{
    const int lane = threadIdx.x & 31;
    const int t    = threadIdx.x >> 5;    // ntile 0..3
    const int lg   = lane & 15;
    const int koff = (lane >= 16) ? 8 : 0;
    const int kk   = blockIdx.x;          // 0..23
    const int mat  = blockIdx.y;          // 0=K,1=Q,2=V

    const float* W = (mat == 0) ? Wk : (mat == 1) ? Wq : Wv;
    const int c0 = kk * 32 + koff;
    const int n  = t * 16 + lg;

    v16bf frag;
    #pragma unroll
    for (int i = 0; i < 8; ++i) {
        frag[i]     = (__bf16)W[(size_t)(c0 + i)      * D_ + n];
        frag[8 + i] = (__bf16)W[(size_t)(c0 + 16 + i) * D_ + n];
    }
    *(v16bf*)(wpk + ((((size_t)mat * KK_ + kk) * 4 + t) * 32 + lane) * 16) = frag;
}

// ---------------------------------------------------------------------------
// Kernel 1: fused projection. Block = 128 thr (4 waves); wave w owns the
// 16-column tile n0 = 16w for one 16-row tile of x and keeps 3 accumulators
// (K, Q, V). One A-fragment load of x feeds 3 WMMAs.
// ---------------------------------------------------------------------------
__global__ __launch_bounds__(128)
void qkv_proj_kernel(const float* __restrict__ x,
                     const __bf16* __restrict__ wpk,
                     __bf16* __restrict__ qws,
                     __bf16* __restrict__ kws,
                     __bf16* __restrict__ vtws)   // V transposed: [D_][B_*T_]
{
    const int lane = threadIdx.x & 31;
    const int wave = threadIdx.x >> 5;
    const int lg   = lane & 15;
    const bool hi  = lane >= 16;
    const int koff = hi ? 8 : 0;
    const int base = hi ? 8 : 0;

    const int rowbase = blockIdx.x * 16;   // row tile in [0, B*T)
    const int n0 = wave * 16;

    v8f accK = {}, accQ = {}, accV = {};

    #pragma unroll 2
    for (int kk = 0; kk < KK_; ++kk) {
        // A fragment: x[rowbase + lg][kk*32 + koff + {0..7, 16..23}]
        v16bf a;
        {
            const float* xa = x + (size_t)(rowbase + lg) * C_ + kk * 32 + koff;
            #pragma unroll
            for (int i = 0; i < 8; ++i) {
                a[i]     = (__bf16)xa[i];
                a[8 + i] = (__bf16)xa[16 + i];
            }
        }
        // Prepacked B fragments: one 32-byte vector load per matrix.
        const __bf16* wb = wpk + (((size_t)kk * 4 + wave) * 32 + lane) * 16;
        const size_t mstride = (size_t)KK_ * 4 * 32 * 16;
        const v16bf bK = *(const v16bf*)(wb);
        const v16bf bQ = *(const v16bf*)(wb + mstride);
        const v16bf bV = *(const v16bf*)(wb + 2 * mstride);

        accK = wmma_bf16(a, bK, accK);
        accQ = wmma_bf16(a, bQ, accQ);
        accV = wmma_bf16(a, bV, accV);
    }

    // K, Q: row-major bf16 [B*T, 64] (attention reads them contiguously).
    {
        __bf16* kp = kws + (size_t)(rowbase + base) * D_ + n0 + lg;
        __bf16* qp = qws + (size_t)(rowbase + base) * D_ + n0 + lg;
        #pragma unroll
        for (int r = 0; r < 8; ++r) {
            kp[(size_t)r * D_] = (__bf16)accK[r];
            qp[(size_t)r * D_] = (__bf16)(accQ[r] * 0.125f);  // 1/sqrt(64)
        }
    }
    // V: transposed store — rows of the C/D fragment are contiguous in V^T,
    // one aligned 16-byte store per lane.
    {
        v8bf o;
        #pragma unroll
        for (int r = 0; r < 8; ++r) o[r] = (__bf16)accV[r];
        *(v8bf*)(vtws + (size_t)(n0 + lg) * (B_ * T_) + rowbase + base) = o;
    }
}

// ---------------------------------------------------------------------------
// Kernel 2: FlashAttention forward. Each wave owns one 16-query tile and the
// full d=64 output. Block = 4 waves = 4 query tiles.
// ---------------------------------------------------------------------------
__global__ __launch_bounds__(128)
void attn_kernel(const __bf16* __restrict__ qws,
                 const __bf16* __restrict__ kws,
                 const __bf16* __restrict__ vtws,
                 float* __restrict__ out)
{
    __shared__ __bf16 Plds[4][16][32];    // per-wave P staging (C -> A layout)

    const int lane = threadIdx.x & 31;
    const int wave = threadIdx.x >> 5;
    const int lg   = lane & 15;
    const bool hi  = lane >= 16;
    const int koff = hi ? 8 : 0;
    const int base = hi ? 8 : 0;

    const int tile  = blockIdx.x * 4 + wave;
    const int b     = tile >> 8;                 // T/16 = 256 tiles per batch
    const int qbase = (tile & 255) << 4;

    // Q fragments (A-layout), d split as c-halves {0..31},{32..63}.
    v16bf Qa[2];
    {
        const __bf16* qb = qws + ((size_t)b * T_ + qbase + lg) * D_;
        #pragma unroll
        for (int h = 0; h < 2; ++h) {
            const __bf16* pq = qb + h * 32 + koff;
            #pragma unroll
            for (int i = 0; i < 8; ++i) {
                Qa[h][i]     = pq[i];
                Qa[h][8 + i] = pq[16 + i];
            }
        }
    }

    v8f acc[4] = {{}, {}, {}, {}};
    float m_i[8], l_i[8];
    #pragma unroll
    for (int r = 0; r < 8; ++r) { m_i[r] = -1e30f; l_i[r] = 0.0f; }

    const int nsteps = (qbase + 16 + 31) >> 5;

    for (int kt = 0; kt < nsteps; ++kt) {
        const int kbase = kt << 5;
        const __bf16* kb = kws + ((size_t)b * T_ + kbase) * D_;

        if (kt + 1 < nsteps)
            __builtin_prefetch(kb + 32 * D_, 0, 1);

        // ---- S = Q K^T, two 16-key subtiles ------------------------------
        v8f S0 = {}, S1 = {};
        #pragma unroll
        for (int h = 0; h < 2; ++h) {
            v16bf Kb;
            const __bf16* pk = kb + (size_t)lg * D_ + h * 32 + koff;
            #pragma unroll
            for (int i = 0; i < 8; ++i) { Kb[i] = pk[i]; Kb[8 + i] = pk[16 + i]; }
            S0 = wmma_bf16(Qa[h], Kb, S0);
        }
        #pragma unroll
        for (int h = 0; h < 2; ++h) {
            v16bf Kb;
            const __bf16* pk = kb + (size_t)(16 + lg) * D_ + h * 32 + koff;
            #pragma unroll
            for (int i = 0; i < 8; ++i) { Kb[i] = pk[i]; Kb[8 + i] = pk[16 + i]; }
            S1 = wmma_bf16(Qa[h], Kb, S1);
        }

        // ---- causal mask (only near the diagonal) ------------------------
        if (kbase + 31 > qbase) {
            #pragma unroll
            for (int r = 0; r < 8; ++r) {
                const int qrow = qbase + base + r;
                if (kbase + lg      > qrow) S0[r] = -1e30f;
                if (kbase + 16 + lg > qrow) S1[r] = -1e30f;
            }
        }

        // ---- online softmax ----------------------------------------------
        float mx[8];
        #pragma unroll
        for (int r = 0; r < 8; ++r) mx[r] = fmaxf(S0[r], S1[r]);
        #pragma unroll
        for (int off = 1; off < 16; off <<= 1)
            #pragma unroll
            for (int r = 0; r < 8; ++r)
                mx[r] = fmaxf(mx[r], __shfl_xor(mx[r], off, 16));

        float al[8], ps[8];
        #pragma unroll
        for (int r = 0; r < 8; ++r) {
            const float mn = fmaxf(m_i[r], mx[r]);
            al[r]  = __expf(m_i[r] - mn);
            m_i[r] = mn;
            const float p0 = __expf(S0[r] - mn);
            const float p1 = __expf(S1[r] - mn);
            S0[r] = p0; S1[r] = p1;
            ps[r] = p0 + p1;
        }
        #pragma unroll
        for (int off = 1; off < 16; off <<= 1)
            #pragma unroll
            for (int r = 0; r < 8; ++r)
                ps[r] += __shfl_xor(ps[r], off, 16);
        #pragma unroll
        for (int r = 0; r < 8; ++r) l_i[r] = l_i[r] * al[r] + ps[r];
        #pragma unroll
        for (int t = 0; t < 4; ++t)
            #pragma unroll
            for (int r = 0; r < 8; ++r) acc[t][r] *= al[r];

        // ---- P: C-layout -> A-layout via per-wave LDS --------------------
        {
            __bf16* pl = &Plds[wave][0][0];
            #pragma unroll
            for (int r = 0; r < 8; ++r) {
                pl[(base + r) * 32 + lg]      = (__bf16)S0[r];
                pl[(base + r) * 32 + 16 + lg] = (__bf16)S1[r];
            }
        }
        asm volatile("s_wait_dscnt 0" ::: "memory");   // same-wave LDS RAW
        v16bf Pa;
        {
            const __bf16* pr = &Plds[wave][lg][koff];
            #pragma unroll
            for (int i = 0; i < 8; ++i) { Pa[i] = pr[i]; Pa[8 + i] = pr[16 + i]; }
        }

        // ---- acc += P V : V^T rows give contiguous key chunks ------------
        #pragma unroll
        for (int t = 0; t < 4; ++t) {
            v16bf Vb;
            const __bf16* pv = vtws + (size_t)(t * 16 + lg) * (B_ * T_)
                                    + (size_t)b * T_ + kbase + koff;
            #pragma unroll
            for (int i = 0; i < 8; ++i) { Vb[i] = pv[i]; Vb[8 + i] = pv[16 + i]; }
            acc[t] = wmma_bf16(Pa, Vb, acc[t]);
        }
    }

    // ---- epilogue: out = acc / l ------------------------------------------
    float* ob = out + ((size_t)b * T_ + qbase + base) * D_;
    #pragma unroll
    for (int r = 0; r < 8; ++r) {
        const float inv = 1.0f / l_i[r];
        #pragma unroll
        for (int t = 0; t < 4; ++t)
            ob[(size_t)r * D_ + t * 16 + lg] = acc[t][r] * inv;
    }
}

// ---------------------------------------------------------------------------
extern "C" void kernel_launch(void* const* d_in, const int* in_sizes, int n_in,
                              void* d_out, int out_size, void* d_ws, size_t ws_size,
                              hipStream_t stream) {
    (void)in_sizes; (void)n_in; (void)out_size; (void)ws_size;

    const float* x  = (const float*)d_in[0];
    const float* Wk = (const float*)d_in[1];
    const float* Wq = (const float*)d_in[2];
    const float* Wv = (const float*)d_in[3];
    float* out = (float*)d_out;

    // Workspace: bf16 Q, K (row-major), V^T, each B*T*D elements (4 MB each),
    // then 288 KB of prepacked W fragments.
    const size_t nt = (size_t)B_ * T_ * D_;
    __bf16* qws  = (__bf16*)d_ws;
    __bf16* kws  = qws + nt;
    __bf16* vtws = kws + nt;
    __bf16* wpk  = vtws + nt;

    // Kernel 0: W prepack (one-shot, tiny).
    dim3 gridW(KK_, 3);
    prep_w_kernel<<<gridW, 128, 0, stream>>>(Wk, Wq, Wv, wpk);

    // Kernel 1: fused Q/K/V projection. 2048 row tiles.
    dim3 gridP((B_ * T_) / 16);
    qkv_proj_kernel<<<gridP, 128, 0, stream>>>(x, wpk, qws, kws, vtws);

    // Kernel 2: attention. 2048 query tiles, 4 per block.
    dim3 gridA((B_ * T_ / 16) / 4);
    attn_kernel<<<gridA, 128, 0, stream>>>(qws, kws, vtws, out);
}
